// LSTMScratch_28501402976812
// MI455X (gfx1250) — matrix operand order repaired and checked by reading, not compiled, not run on hardware
//
#include <hip/hip_runtime.h>

// ---------------------------------------------------------------------------
// LSTM cell for MI455X (gfx1250): fused dual-GEMM (x@Wx.T + h@Wh.T) as one
// bf16 WMMA GEMM [16384,1024]x[1024,2048] with f32 accumulation, double-
// buffered LDS staging via GLOBAL_LOAD_ASYNC_TO_LDS_B128 (ASYNCcnt), and the
// LSTM nonlinearity fused into the epilogue (gates never touch memory).
// ---------------------------------------------------------------------------

#define BDIM 16384
#define EDIM 512
#define HDIM 512
#define KDIM 1024      // E + H fused K
#define NDIM 2048      // 4*H fused N

#define BM 64          // M rows per block
#define BN 32          // H columns per block (x4 gates -> 128 N columns)
#define BK 64          // K per LDS stage
#define NKSTAGES (KDIM / BK)
#define ASTR 72        // LDS row stride (halves): 144B, 16B-aligned, padded
#define BSTR 72

typedef __bf16 bf16_t;
typedef __bf16 v16bf __attribute__((ext_vector_type(16)));
typedef __bf16 v8bf  __attribute__((ext_vector_type(8)));
typedef float  v8f   __attribute__((ext_vector_type(8)));
typedef int    v4i   __attribute__((ext_vector_type(4)));

#if defined(__has_builtin)
#if __has_builtin(__builtin_amdgcn_global_load_async_to_lds_b128)
#define USE_ASYNC_LDS 1
#endif
#endif
#ifndef USE_ASYNC_LDS
#define USE_ASYNC_LDS 0
#endif

// builtin signature (from hipcc diagnostic): param0 = v4i __device__* (AS1),
// param1 = LDS v4i* (AS3), then i32 offset, i32 cpol.
#define AS_GLOBAL_V4I(p) ((__attribute__((address_space(1))) v4i*)((void*)(p)))
#define AS_LDS_V4I(p)    ((__attribute__((address_space(3))) v4i*)((void*)(p)))

__device__ __forceinline__ void wait_async_all() {
#if USE_ASYNC_LDS
#if __has_builtin(__builtin_amdgcn_s_wait_asynccnt)
  __builtin_amdgcn_s_wait_asynccnt(0);
#else
  asm volatile("s_wait_asynccnt 0x0" ::: "memory");
#endif
#endif
}

__device__ __forceinline__ unsigned short f32_to_bf16_bits(float f) {
  unsigned int u = __float_as_uint(f);
  u += 0x7FFFu + ((u >> 16) & 1u);   // round-to-nearest-even
  return (unsigned short)(u >> 16);
}

__device__ __forceinline__ float sigmoid_fast(float z) {
  return 1.0f / (1.0f + __expf(-z));   // z->-inf: exp->inf, 1/inf -> 0 (safe)
}

__device__ __forceinline__ float tanh_fast(float z) {
  // branchless, overflow-safe: t in (0,1]
  float t = __expf(-2.0f * __builtin_fabsf(z));
  float r = (1.0f - t) / (1.0f + t);
  return __builtin_copysignf(r, z);
}

// ---------------------------------------------------------------------------
// Pack A = [bf16(x) | bf16(h_prev)]  ->  A4 [16384, 1024]
// ---------------------------------------------------------------------------
__global__ __launch_bounds__(256) void lstm_pack_A(
    const float* __restrict__ x, const float* __restrict__ h_prev,
    unsigned short* __restrict__ A4) {
  size_t t = (size_t)blockIdx.x * 256 + threadIdx.x;   // over 16384*512
  size_t m = t >> 9;
  size_t k = t & 511;
  A4[(m << 10) + k]        = f32_to_bf16_bits(x[t]);
  A4[(m << 10) + 512 + k]  = f32_to_bf16_bits(h_prev[t]);
}

// ---------------------------------------------------------------------------
// Pack W4 [2048, 1024]: row gi*512+n = [bf16(Wx_gi[n,:]) | bf16(Wh_gi[n,:])]
// ---------------------------------------------------------------------------
__global__ __launch_bounds__(256) void lstm_pack_W(
    const float* __restrict__ Wx0, const float* __restrict__ Wh0,
    const float* __restrict__ Wx1, const float* __restrict__ Wh1,
    const float* __restrict__ Wx2, const float* __restrict__ Wh2,
    const float* __restrict__ Wx3, const float* __restrict__ Wh3,
    unsigned short* __restrict__ W4) {
  size_t t = (size_t)blockIdx.x * 256 + threadIdx.x;   // over 4*512*512
  int    gi  = (int)(t >> 18);
  size_t rem = t & ((size_t)(1 << 18) - 1);
  size_t n   = rem >> 9;
  size_t k   = rem & 511;
  const float* Wx = (gi == 0) ? Wx0 : (gi == 1) ? Wx1 : (gi == 2) ? Wx2 : Wx3;
  const float* Wh = (gi == 0) ? Wh0 : (gi == 1) ? Wh1 : (gi == 2) ? Wh2 : Wh3;
  size_t row = ((size_t)gi << 9) + n;
  W4[(row << 10) + k]       = f32_to_bf16_bits(Wx[(n << 9) + k]);
  W4[(row << 10) + 512 + k] = f32_to_bf16_bits(Wh[(n << 9) + k]);
}

__global__ __launch_bounds__(256) void lstm_pack_bias(
    const float* __restrict__ bx0, const float* __restrict__ bh0,
    const float* __restrict__ bx1, const float* __restrict__ bh1,
    const float* __restrict__ bx2, const float* __restrict__ bh2,
    const float* __restrict__ bx3, const float* __restrict__ bh3,
    float* __restrict__ bias) {
  int t  = blockIdx.x * 256 + threadIdx.x;   // over 2048
  int gi = t >> 9;
  int n  = t & 511;
  const float* bx = (gi == 0) ? bx0 : (gi == 1) ? bx1 : (gi == 2) ? bx2 : bx3;
  const float* bh = (gi == 0) ? bh0 : (gi == 1) ? bh1 : (gi == 2) ? bh2 : bh3;
  bias[t] = bx[n] + bh[n];
}

// ---------------------------------------------------------------------------
// Stage one BK-slice of A (64x64) and B (128x64) into an LDS buffer.
// Async path: GLOBAL_LOAD_ASYNC_TO_LDS_B128 (no VGPR bounce, ASYNCcnt).
// ---------------------------------------------------------------------------
__device__ __forceinline__ void stage_tiles(
    bf16_t* __restrict__ As, bf16_t* __restrict__ Bs,
    const unsigned short* __restrict__ A4,
    const unsigned short* __restrict__ W4,
    int row0, int col0, int kb, int tid) {
#pragma unroll
  for (int i = 0; i < 2; ++i) {          // A: 512 x 16B chunks
    int c = tid + i * 256;
    int r = c >> 3, c8 = c & 7;
    const unsigned short* g = &A4[(size_t)(row0 + r) * KDIM + kb * BK + c8 * 8];
    bf16_t* l = &As[r * ASTR + c8 * 8];
#if USE_ASYNC_LDS
    __builtin_amdgcn_global_load_async_to_lds_b128(AS_GLOBAL_V4I(g),
                                                   AS_LDS_V4I(l), 0, 0);
#else
    *(v8bf*)l = *(const v8bf*)g;
#endif
  }
#pragma unroll
  for (int i = 0; i < 4; ++i) {          // B: 1024 x 16B chunks
    int c  = tid + i * 256;
    int rb = c >> 3, c8 = c & 7;
    int gi = rb >> 5, j = rb & 31;
    const unsigned short* g =
        &W4[(size_t)(gi * HDIM + col0 + j) * KDIM + kb * BK + c8 * 8];
    bf16_t* l = &Bs[rb * BSTR + c8 * 8];
#if USE_ASYNC_LDS
    __builtin_amdgcn_global_load_async_to_lds_b128(AS_GLOBAL_V4I(g),
                                                   AS_LDS_V4I(l), 0, 0);
#else
    *(v8bf*)l = *(const v8bf*)g;
#endif
  }
}

// ---------------------------------------------------------------------------
// Fused GEMM + LSTM epilogue. Block: 256 threads (8 waves).
// Output tile: 64 rows x 32 h-cols x 4 gates; wave (wave_m, wave_c) owns all
// four gate tiles for its 16x16 (m, hcol) tile -> wave-local epilogue.
// Double-buffered LDS: prefetch stage kb+1 (async) while computing kb.
// ---------------------------------------------------------------------------
__global__ __launch_bounds__(256) void lstm_gemm_fused(
    const unsigned short* __restrict__ A4,   // [16384,1024] bf16 bits
    const unsigned short* __restrict__ W4,   // [2048, 1024] bf16 bits
    const float* __restrict__ bias,          // [2048]
    const float* __restrict__ c_prev,        // [16384,512]
    float* __restrict__ h_out,               // [16384,512]
    float* __restrict__ c_out) {             // [16384,512]
  __shared__ bf16_t As[2][BM * ASTR];        // 2 x  64 x 64 halves (padded)
  __shared__ bf16_t Bs[2][4 * BN * BSTR];    // 2 x 128 x 64 halves (padded)

  const int tid    = threadIdx.x;
  const int lane   = tid & 31;
  const int w      = tid >> 5;
  const int wave_m = w & 3;
  const int wave_c = w >> 2;
  const int sel    = lane >> 4;          // half-wave select (CDNA5 frag layout)
  const int lan16  = lane & 15;
  const int row0   = blockIdx.y * BM;    // batch-row base
  const int col0   = blockIdx.x * BN;    // h-column base

  v8f acc[4] = {};                       // one 16x16 f32 tile per gate

  // prologue: fill buffer 0
  stage_tiles(As[0], Bs[0], A4, W4, row0, col0, 0, tid);
  wait_async_all();
  __syncthreads();

  for (int kb = 0; kb < NKSTAGES; ++kb) {
    const int cur = kb & 1;
    if (kb + 1 < NKSTAGES)
      stage_tiles(As[cur ^ 1], Bs[cur ^ 1], A4, W4, row0, col0, kb + 1, tid);

    // ---- load ALL fragments for this stage, then 8 back-to-back WMMAs ----
    const bf16_t* abase = &As[cur][(wave_m * 16 + lan16) * ASTR];
    const bf16_t* bbase = &Bs[cur][(wave_c * 16 + lan16) * BSTR];

    union F { v16bf v; v8bf h[2]; };
    F a[2];
#pragma unroll
    for (int t = 0; t < 2; ++t) {
      // A 16x32 MxK: lane<16 holds K 0-7 & 16-23; lane>=16 holds K 8-15 & 24-31
      const int kk = t * 32;
      a[t].h[0] = *(const v8bf*)&abase[kk + sel * 8];
      a[t].h[1] = *(const v8bf*)&abase[kk + 16 + sel * 8];
    }
    F b[4][2];
#pragma unroll
    for (int gi = 0; gi < 4; ++gi) {
      const bf16_t* br = bbase + gi * 32 * BSTR;
#pragma unroll
      for (int t = 0; t < 2; ++t) {
        // B 32x16 KxN: lane<16 holds K 0-7 & 16-23; lane>=16 holds K 8-15 & 24-31
        const int kk = t * 32;
        b[gi][t].h[0] = *(const v8bf*)&br[kk + sel * 8];
        b[gi][t].h[1] = *(const v8bf*)&br[kk + 16 + sel * 8];
      }
    }
#pragma unroll
    for (int t = 0; t < 2; ++t)
#pragma unroll
      for (int gi = 0; gi < 4; ++gi)
        acc[gi] = __builtin_amdgcn_wmma_f32_16x16x32_bf16(
            false, a[t].v, false, b[gi][t].v, (short)0, acc[gi], false, false);

    wait_async_all();       // next-stage tiles landed in the other buffer
    __syncthreads();        // all waves done reading cur + writing nxt
  }

  // ---- epilogue: all four gates for this wave's 16x16 tile are local ----
  const int   hcol  = col0 + wave_c * 16 + lan16;
  const float bi    = bias[hcol];
  const float bff   = bias[512 + hcol];
  const float bg    = bias[1024 + hcol];
  const float bo    = bias[1536 + hcol];
  const int   mbase = row0 + wave_m * 16 + sel * 8;   // C/D layout: lane>=16 -> M+8

#pragma unroll
  for (int v = 0; v < 8; ++v) {
    const int    m   = mbase + v;
    const size_t idx = (size_t)m * HDIM + hcol;
    float ig = sigmoid_fast(acc[0][v] + bi);
    float fg = sigmoid_fast(acc[1][v] + bff);
    float gg = tanh_fast(acc[2][v] + bg);
    float og = sigmoid_fast(acc[3][v] + bo);
    float cn = fg * c_prev[idx] + ig * gg;
    float hn = og * tanh_fast(cn);
    h_out[idx] = hn;
    c_out[idx] = cn;
  }
}

// ---------------------------------------------------------------------------
extern "C" void kernel_launch(void* const* d_in, const int* in_sizes, int n_in,
                              void* d_out, int out_size, void* d_ws, size_t ws_size,
                              hipStream_t stream) {
  (void)in_sizes; (void)n_in; (void)out_size; (void)ws_size;

  const float* x      = (const float*)d_in[0];
  const float* h_prev = (const float*)d_in[1];
  const float* c_prev = (const float*)d_in[2];
  // dict order: per gate g in {i,f,g,o}: Wx(3+4g), bx(4+4g), Wh(5+4g), bh(6+4g)

  unsigned short* A4 = (unsigned short*)d_ws;                          // 32 MB
  unsigned short* W4 =
      (unsigned short*)((char*)d_ws + (size_t)BDIM * KDIM * 2);        // 4 MB
  float* bias =
      (float*)((char*)d_ws + (size_t)BDIM * KDIM * 2 + (size_t)NDIM * KDIM * 2);

  float* h_out = (float*)d_out;
  float* c_out = h_out + (size_t)BDIM * HDIM;

  lstm_pack_A<<<(BDIM * EDIM) / 256, 256, 0, stream>>>(x, h_prev, A4);

  lstm_pack_W<<<(4 * 512 * 512) / 256, 256, 0, stream>>>(
      (const float*)d_in[3],  (const float*)d_in[5],
      (const float*)d_in[7],  (const float*)d_in[9],
      (const float*)d_in[11], (const float*)d_in[13],
      (const float*)d_in[15], (const float*)d_in[17], W4);

  lstm_pack_bias<<<NDIM / 256, 256, 0, stream>>>(
      (const float*)d_in[4],  (const float*)d_in[6],
      (const float*)d_in[8],  (const float*)d_in[10],
      (const float*)d_in[12], (const float*)d_in[14],
      (const float*)d_in[16], (const float*)d_in[18], bias);

  dim3 grid(HDIM / BN, BDIM / BM);   // 16 x 256 blocks
  lstm_gemm_fused<<<grid, 256, 0, stream>>>(A4, W4, bias, c_prev, h_out, c_out);
}
